// IntraSA_76089640616035
// MI455X (gfx1250) — matrix operand order repaired and checked by reading, not compiled, not run on hardware
//
#include <hip/hip_runtime.h>
#include <hip/hip_bf16.h>

typedef _Float16 h16;
typedef __attribute__((ext_vector_type(16))) _Float16 v16h;
typedef __attribute__((ext_vector_type(8)))  _Float16 v8h;
typedef __attribute__((ext_vector_type(8)))  float    v8f;
typedef __attribute__((ext_vector_type(4)))  int      v4i;

// --------------------------------------------------------------------------
// CDNA5 async global->LDS DMA (verified working on this toolchain)
// --------------------------------------------------------------------------
#if __has_builtin(__builtin_amdgcn_global_load_async_to_lds_b128) && \
    __has_builtin(__builtin_amdgcn_s_wait_asynccnt)
#define USE_ASYNC_LDS 1
#else
#define USE_ASYNC_LDS 0
#endif

#if USE_ASYNC_LDS
#define ASYNC_CP16(dst, src)                                                  \
    __builtin_amdgcn_global_load_async_to_lds_b128(                           \
        (__attribute__((address_space(1))) v4i*)(src),                        \
        (__attribute__((address_space(3))) v4i*)(dst), 0, 0)
#endif

// ---------------------------------------------------------------------------
// CDNA5 WMMA helpers (wave32, V_WMMA_F32_16X16X32_F16)
// A 16x32 f16, B 32x16 f16 (stored transposed: [N][K], K contiguous), C/D f32
// ---------------------------------------------------------------------------
__device__ __forceinline__ v16h mk16(v8h lo, v8h hi) {
    v16h r;
#pragma unroll
    for (int j = 0; j < 8; ++j) { r[j] = lo[j]; r[j + 8] = hi[j]; }
    return r;
}

__device__ __forceinline__ v16h frag_a(const h16* base, int row0, int k0, int ld) {
    const int lane = threadIdx.x & 31;
    const int m = lane & 15, g = lane >> 4;
    const h16* p = base + (size_t)(row0 + m) * ld + k0 + 8 * g;
    return mk16(*(const v8h*)p, *(const v8h*)(p + 16));
}

__device__ __forceinline__ v16h frag_b(const h16* base, int n0, int k0, int ld) {
    const int lane = threadIdx.x & 31;
    const int n = lane & 15, g = lane >> 4;
    const h16* p = base + (size_t)(n0 + n) * ld + k0 + 8 * g;
    return mk16(*(const v8h*)p, *(const v8h*)(p + 16));
}

__device__ __forceinline__ v8f wmma_f(v16h a, v16h b, v8f c) {
    return __builtin_amdgcn_wmma_f32_16x16x32_f16(false, a, false, b,
                                                  (short)0, c, false, false);
}

// ---------------------------------------------------------------------------
// Kernel 0: f32 -> f16 weight conversion
// ---------------------------------------------------------------------------
__global__ void k_cvt(const float* __restrict__ src, h16* __restrict__ dst, int n) {
    int i = blockIdx.x * 256 + threadIdx.x;
    if (i < n) dst[i] = (h16)src[i];
}

// ---------------------------------------------------------------------------
// Stage one conv-weight K-panel (160 rows x 32 k-halves = 10 KB) into LDS.
// Async DMA when available; 2 x 16B chunks per thread (320 threads).
// ---------------------------------------------------------------------------
__device__ __forceinline__ void stage_panel(const h16* __restrict__ wc, h16* buf,
                                            int half, int kt, int tid) {
    const h16* srcp = wc + (size_t)(half * 160) * 320 + 32 * kt;
#pragma unroll
    for (int it = 0; it < 2; ++it) {
        const int idx = tid + it * 320;
        const int nrow = idx >> 2, j = idx & 3;
#if USE_ASYNC_LDS
        ASYNC_CP16((uint4*)(buf + nrow * 32) + j,
                   (const uint4*)(srcp + (size_t)nrow * 320) + j);
#else
        ((uint4*)(buf + nrow * 32))[j] = ((const uint4*)(srcp + (size_t)nrow * 320))[j];
#endif
    }
}

// ---------------------------------------------------------------------------
// Kernel 1: per-pixel LayerNorm + 1x1 conv GEMM (160 px x 320 x 320 per block)
// Block: 320 threads = 10 waves; one (b,h) image row per block.
// Weight K-panels double-buffered in LDS via async DMA; buffer selection is
// compile-time (kt loop fully unrolled) so B-fragments stay on the DS path.
// ---------------------------------------------------------------------------
__global__ void __launch_bounds__(320)
k_lnconv(const float* __restrict__ x, const float* __restrict__ lng,
         const float* __restrict__ lnb, const h16* __restrict__ wc,
         const float* __restrict__ cb, h16* __restrict__ xh, h16* __restrict__ xv)
{
    extern __shared__ h16 sm[];
    h16* Xs  = sm;                        // [160][320] f16 = 102400 B
    h16* Wp0 = sm + 51200;                // K-panel buf 0: 160x32 = 10240 B
    h16* Wp1 = sm + 56320;                // K-panel buf 1
    float* red = (float*)(sm + 61440);    // 640 f32 reduction scratch

    const int blk = blockIdx.x;           // 0..1279  -> (b, hh)
    const int bb = blk / 160, hh = blk % 160;
    const int tid = threadIdx.x;

    // Kick off panel (half 0, kt 0) DMA before LN: copies land while ALUs work.
    stage_panel(wc, Wp0, 0, 0, tid);

    // ---- LayerNorm: 2 threads per pixel (interleaved channels) ----
    {
        const int p = tid >> 1, hf = tid & 1;
        const float* px = x + (size_t)bb * 320 * 25600 + (size_t)hh * 160 + p;
        float s = 0.f, s2 = 0.f;
        for (int i = 0; i < 160; ++i) {
            const int c = 2 * i + hf;
            float v = px[(size_t)c * 25600];
            s += v; s2 += v * v;
            Xs[p * 320 + c] = (h16)v;
        }
        red[tid] = s; red[320 + tid] = s2;
        __syncthreads();
        if (hf == 0) {
            const float st  = red[2 * p] + red[2 * p + 1];
            const float s2t = red[320 + 2 * p] + red[320 + 2 * p + 1];
            const float mu  = st * (1.f / 320.f);
            const float var = s2t * (1.f / 320.f) - mu * mu;
            red[2 * p] = mu; red[2 * p + 1] = rsqrtf(var + 1e-5f);
        }
        __syncthreads();
        const float mu = red[2 * p], rs = red[2 * p + 1];
        for (int i = 0; i < 160; ++i) {
            const int c = 2 * i + hf;
            float v = ((float)Xs[p * 320 + c] - mu) * rs * lng[c] + lnb[c];
            Xs[p * 320 + c] = (h16)v;
        }
    }

    // ---- GEMM: wave w -> 16-pixel M tile; 20 N tiles in 2 half passes ----
    const int wv = tid >> 5;
    const int lane = tid & 31, n = lane & 15, g = lane >> 4;
    const int p0 = blk * 160;

#pragma unroll
    for (int half = 0; half < 2; ++half) {
        const int nb = half * 10;
        v8f acc[10];
#pragma unroll
        for (int t = 0; t < 10; ++t) acc[t] = v8f{};

#pragma unroll
        for (int kt = 0; kt < 10; ++kt) {
            __syncthreads();   // everyone done reading the buffer being restaged
            if (kt < 9)          stage_panel(wc, (kt & 1) ? Wp0 : Wp1, half, kt + 1, tid);
            else if (half == 0)  stage_panel(wc, Wp0, 1, 0, tid);   // prefetch next half
#if USE_ASYNC_LDS
            if (kt < 9 || half == 0) __builtin_amdgcn_s_wait_asynccnt(2); // panel kt done
            else                     __builtin_amdgcn_s_wait_asynccnt(0);
#endif
            __syncthreads();   // panel kt visible to all waves

            v16h a = frag_a(Xs, 16 * wv, 32 * kt, 320);
            const h16* bp = (kt & 1) ? Wp1 : Wp0;   // compile-time after unroll
#pragma unroll
            for (int nt = 0; nt < 10; ++nt)
                acc[nt] = wmma_f(a, frag_b(bp, 16 * nt, 0, 32), acc[nt]);
        }

#pragma unroll
        for (int nt = 0; nt < 10; ++nt) {
            const int ch = 16 * (nb + nt) + n;
            const float bias = cb[ch];
#pragma unroll
            for (int i = 0; i < 8; ++i) {
                const int m = 16 * wv + i + 8 * g;          // ww
                const h16 v = (h16)(acc[nt][i] + bias);
                if (ch < 160) xh[(size_t)(p0 + m) * 160 + ch] = v;
                else xv[(((size_t)bb * 160 + m) * 160 + hh) * 160 + (ch - 160)] = v;
            }
        }
    }
}

// ---------------------------------------------------------------------------
// Kernel 2: fused QKV + MHA for one (row, head, branch).
// Block: 320 threads = 10 waves; seq=160 (10 M tiles), head-dim 32 (2 N tiles).
// ---------------------------------------------------------------------------
__global__ void __launch_bounds__(320)
k_attn(const h16* __restrict__ xh, const h16* __restrict__ xv,
       const h16* __restrict__ wqh, const h16* __restrict__ wkh, const h16* __restrict__ wvh,
       const h16* __restrict__ wqv, const h16* __restrict__ wkv, const h16* __restrict__ wvv,
       const float* __restrict__ bqh, const float* __restrict__ bkh, const float* __restrict__ bvh,
       const float* __restrict__ bqv, const float* __restrict__ bkv, const float* __restrict__ bvv,
       float* __restrict__ out)
{
    const int row = blockIdx.x, head = blockIdx.y, br = blockIdx.z;
    const h16* X   = (br ? xv : xh) + (size_t)row * 25600;
    const h16* wq  = (br ? wqv : wqh) + head * 32 * 160;   // [32][160], K contiguous
    const h16* wk  = (br ? wkv : wkh) + head * 32 * 160;
    const h16* wvm = (br ? wvv : wvh) + head * 32 * 160;
    const float* bq = (br ? bqv : bqh) + head * 32;
    const float* bk = (br ? bkv : bkh) + head * 32;
    const float* bv = (br ? bvv : bvh) + head * 32;

    extern __shared__ h16 smem[];
    h16* Xp = smem;                  // 25600 halves: X, later P (aliased)
    h16* Ws = smem + 25600;          // 3 * 5120 halves: Wq|Wk|Wv slices
    h16* Qs = smem + 25600 + 15360;  // 160x32
    h16* Ks = Qs + 5120;             // 160x32
    h16* Vt = Ks + 5120;             // 32x160 (transposed V)

    const int tid = threadIdx.x;
    {   // ---- stage X row + weight slices into LDS via async DMA ----
        const uint4* s  = (const uint4*)X;    uint4* d  = (uint4*)Xp;
        const uint4* sq = (const uint4*)wq;   uint4* dq = (uint4*)Ws;
        const uint4* sk = (const uint4*)wk;   uint4* dk = (uint4*)(Ws + 5120);
        const uint4* sv = (const uint4*)wvm;  uint4* dv = (uint4*)(Ws + 10240);
#if USE_ASYNC_LDS
        for (int i = tid; i < 3200; i += 320) ASYNC_CP16(d + i, s + i);
        for (int i = tid; i < 640; i += 320) {
            ASYNC_CP16(dq + i, sq + i);
            ASYNC_CP16(dk + i, sk + i);
            ASYNC_CP16(dv + i, sv + i);
        }
        __builtin_amdgcn_s_wait_asynccnt(0);
#else
        for (int i = tid; i < 3200; i += 320) d[i] = s[i];
        for (int i = tid; i < 640; i += 320) { dq[i] = sq[i]; dk[i] = sk[i]; dv[i] = sv[i]; }
#endif
    }
    __syncthreads();

    const int wvid = tid >> 5;
    const int lane = tid & 31, n = lane & 15, g = lane >> 4;
    const float scale = 0.17677669529663687f;     // 1/sqrt(32), folded into Q

    // ---- Q = X Wq^T, K = X Wk^T, V = X Wv^T (per-head 160x32 GEMMs) ----
#pragma unroll
    for (int mat = 0; mat < 3; ++mat) {
        const h16* Wm = Ws + mat * 5120;
        v8f a0 = v8f{}, a1 = v8f{};
#pragma unroll
        for (int kt = 0; kt < 5; ++kt) {
            v16h af = frag_a(Xp, 16 * wvid, 32 * kt, 160);
            a0 = wmma_f(af, frag_b(Wm, 0, 32 * kt, 160), a0);
            a1 = wmma_f(af, frag_b(Wm, 16, 32 * kt, 160), a1);
        }
        const float* bias = (mat == 0) ? bq : (mat == 1) ? bk : bv;
#pragma unroll
        for (int nt = 0; nt < 2; ++nt) {
            v8f& A = nt ? a1 : a0;
            const int c = nt * 16 + n;
            const float bb_ = bias[c];
#pragma unroll
            for (int i = 0; i < 8; ++i) {
                const int m = 16 * wvid + i + 8 * g;
                const float val = A[i] + bb_;
                if (mat == 0)      Qs[m * 32 + c] = (h16)(val * scale);
                else if (mat == 1) Ks[m * 32 + c] = (h16)val;
                else               Vt[c * 160 + m] = (h16)val;   // store V transposed
            }
        }
    }
    __syncthreads();

    // ---- S = Q K^T : one WMMA per 16x16 tile (K = 32 = head dim) ----
    v8f S[10];
    {
        v16h qa = frag_a(Qs, 16 * wvid, 0, 32);
#pragma unroll
        for (int t = 0; t < 10; ++t)
            S[t] = wmma_f(qa, frag_b(Ks, 16 * t, 0, 32), v8f{});
    }

    // ---- softmax over 160 cols; rows live in 16-lane groups (wave32) ----
#pragma unroll
    for (int i = 0; i < 8; ++i) {
        float mx = -1e30f;
#pragma unroll
        for (int t = 0; t < 10; ++t) mx = fmaxf(mx, S[t][i]);
        for (int off = 8; off; off >>= 1) mx = fmaxf(mx, __shfl_xor(mx, off));
        float sum = 0.f;
#pragma unroll
        for (int t = 0; t < 10; ++t) { float e = __expf(S[t][i] - mx); S[t][i] = e; sum += e; }
        for (int off = 8; off; off >>= 1) sum += __shfl_xor(sum, off);
        const float r = 1.f / sum;
        const int m = 16 * wvid + i + 8 * g;
#pragma unroll
        for (int t = 0; t < 10; ++t) Xp[m * 160 + t * 16 + n] = (h16)(S[t][i] * r);
    }
    __syncthreads();

    // ---- O = P V ----
    v8f o0 = v8f{}, o1 = v8f{};
#pragma unroll
    for (int kt = 0; kt < 5; ++kt) {
        v16h pa = frag_a(Xp, 16 * wvid, 32 * kt, 160);
        o0 = wmma_f(pa, frag_b(Vt, 0, 32 * kt, 160), o0);
        o1 = wmma_f(pa, frag_b(Vt, 16, 32 * kt, 160), o1);
    }

    // ---- scatter to out[(br*8+b), ch, h, w] (f32) ----
    const int bb = row / 160, rr = row % 160;
#pragma unroll
    for (int nt = 0; nt < 2; ++nt) {
        v8f& O = nt ? o1 : o0;
        const int ch = head * 32 + nt * 16 + n;
#pragma unroll
        for (int i = 0; i < 8; ++i) {
            const int m = 16 * wvid + i + 8 * g;     // seq position
            const int hh = br ? m : rr;
            const int ww = br ? rr : m;
            const int ob = br * 8 + bb;
            out[((size_t)ob * 160 + ch) * 25600 + hh * 160 + ww] = O[i];
        }
    }
}

// ---------------------------------------------------------------------------
extern "C" void kernel_launch(void* const* d_in, const int* in_sizes, int n_in,
                              void* d_out, int out_size, void* d_ws, size_t ws_size,
                              hipStream_t stream) {
    const float* x      = (const float*)d_in[0];
    const float* ln_g   = (const float*)d_in[1];
    const float* ln_b   = (const float*)d_in[2];
    const float* conv_w = (const float*)d_in[3];
    const float* conv_b = (const float*)d_in[4];

    h16* ws16 = (h16*)d_ws;
    h16* wc16  = ws16;                   // 320*320
    h16* wqh16 = ws16 + 102400;          // each 160*160
    h16* wkh16 = ws16 + 128000;
    h16* wvh16 = ws16 + 153600;
    h16* wqv16 = ws16 + 179200;
    h16* wkv16 = ws16 + 204800;
    h16* wvv16 = ws16 + 230400;
    h16* xh    = ws16 + 256000;          // 204800 * 160 f16
    h16* xv    = xh + (size_t)204800 * 160;

    k_cvt<<<(102400 + 255) / 256, 256, 0, stream>>>(conv_w, wc16, 102400);
    k_cvt<<<(25600 + 255) / 256, 256, 0, stream>>>((const float*)d_in[5],  wqh16, 25600);
    k_cvt<<<(25600 + 255) / 256, 256, 0, stream>>>((const float*)d_in[7],  wkh16, 25600);
    k_cvt<<<(25600 + 255) / 256, 256, 0, stream>>>((const float*)d_in[9],  wvh16, 25600);
    k_cvt<<<(25600 + 255) / 256, 256, 0, stream>>>((const float*)d_in[11], wqv16, 25600);
    k_cvt<<<(25600 + 255) / 256, 256, 0, stream>>>((const float*)d_in[13], wkv16, 25600);
    k_cvt<<<(25600 + 255) / 256, 256, 0, stream>>>((const float*)d_in[15], wvv16, 25600);

    // LN + 1x1 conv: 1280 blocks; LDS = X panel + 2 weight panels + reduction
    k_lnconv<<<1280, 320, (51200 + 2 * 5120) * 2 + 640 * 4, stream>>>(
        x, ln_g, ln_b, wc16, conv_b, xh, xv);

    // Fused attention: (row, head, branch), 110 KB LDS
    k_attn<<<dim3(1280, 5, 2), 320, (25600 + 15360 + 3 * 5120) * 2, stream>>>(
        xh, xv, wqh16, wkh16, wvh16, wqv16, wkv16, wvv16,
        (const float*)d_in[6], (const float*)d_in[8], (const float*)d_in[10],
        (const float*)d_in[12], (const float*)d_in[14], (const float*)d_in[16],
        (float*)d_out);
}
